// LSTM_33595234189617
// MI455X (gfx1250) — matrix operand order repaired and checked by reading, not compiled, not run on hardware
//
#include <hip/hip_runtime.h>
#include <hip/hip_bf16.h>

#define T_STEPS 512
#define H 50

typedef __attribute__((ext_vector_type(16))) __bf16 v16bf;
typedef __attribute__((ext_vector_type(8)))  float  v8f;

union BfPack {
  v16bf v;
  unsigned short s[16];
  uint4 q[2];
};

__device__ __forceinline__ unsigned short f2bf(float f) {
  unsigned int u = __float_as_uint(f);
  u += 0x7FFFu + ((u >> 16) & 1u);          // round-to-nearest-even
  return (unsigned short)(u >> 16);
}
__device__ __forceinline__ float bf2f(unsigned short s) {
  return __uint_as_float(((unsigned int)s) << 16);
}
__device__ __forceinline__ float fsigmoid(float x) {
  return __builtin_amdgcn_rcpf(1.0f + __expf(-x));
}
__device__ __forceinline__ float ftanh(float x) {
  float e = __expf(-2.0f * x);
  return (1.0f - e) * __builtin_amdgcn_rcpf(1.0f + e);
}

__global__ __launch_bounds__(128, 1) void lstm_wmma_kernel(
    const float* __restrict__ seq,  const float* __restrict__ W_ih,
    const float* __restrict__ W_hh, const float* __restrict__ b_ih,
    const float* __restrict__ b_hh, const float* __restrict__ W_out,
    const float* __restrict__ b_out, float* __restrict__ out) {

  // per-wave 16x64 bf16 staging tile: [h(0..49) | zeros | x(56..59) | zeros]
  __shared__ __align__(16) unsigned short Abuf[4][16][64];

  const int lane = threadIdx.x & 31;
  const int w    = threadIdx.x >> 5;
  const int half = lane >> 4;
  const int n15  = lane & 15;
  const int m0   = (blockIdx.x * 4 + w) * 16;   // batch-row base of this wave

  // zero staging (h0 = 0 + static zero padding in K=50..55,60..63)
  for (int i = lane; i < 16 * 64; i += 32)
    (&Abuf[w][0][0])[i] = 0;

  // ---- resident weights: 16 N-tiles x 2 K-tiles in B-operand layout ----
  // N-tile j: gate = j>>2, gate-cols 16*(j&3)..+15 (cols >=50 are zero pad)
  // K = 0..49 -> W_hh^T, K = 56..59 -> W_ih^T, else 0
  v16bf Bop[16][2];
  float biasv[16];
  #pragma unroll
  for (int j = 0; j < 16; ++j) {
    const int  gate = j >> 2;
    const int  cig  = 16 * (j & 3) + n15;
    const bool cv   = (cig < H);
    const int  wrow = gate * H + (cv ? cig : 0);
    biasv[j] = cv ? (b_ih[wrow] + b_hh[wrow]) : 0.0f;
    #pragma unroll
    for (int kt = 0; kt < 2; ++kt) {
      BfPack t;
      #pragma unroll
      for (int e = 0; e < 16; ++e) {
        const int K = 32 * kt + 16 * half + e;   // B layout: lane-half picks K group
        float wv = 0.0f;
        if (cv) {
          if (K < H)                  wv = W_hh[wrow * H + K];
          else if (K >= 56 && K < 60) wv = W_ih[wrow * 4 + (K - 56)];
        }
        t.s[e] = f2bf(wv);
      }
      Bop[j][kt] = t.v;
    }
  }

  // cell state c for 4 column-subtiles, D-layout, f32, resident for all T
  v8f cs[4];
  #pragma unroll
  for (int s = 0; s < 4; ++s) cs[s] = (v8f)0.0f;

  const float4* seq4 = (const float4*)seq;

  #pragma unroll 1
  for (int t = 0; t < T_STEPS; ++t) {
    // inject x_t into staging cols 56..59 (bf16 pairs, 8B-aligned ds_store)
    if (lane < 16) {
      float4 x = seq4[(size_t)(m0 + lane) * T_STEPS + t];
      uint2 p;
      p.x = (unsigned)f2bf(x.x) | ((unsigned)f2bf(x.y) << 16);
      p.y = (unsigned)f2bf(x.z) | ((unsigned)f2bf(x.w) << 16);
      *reinterpret_cast<uint2*>(&Abuf[w][lane][56]) = p;
    }
    __builtin_amdgcn_wave_barrier();

    // A operands (16x32 bf16 each): per lane two contiguous 16B chunks
    BfPack a0, a1;
    {
      const int m = n15;
      a0.q[0] = *reinterpret_cast<const uint4*>(&Abuf[w][m][ 0 + 16 * half]);
      a0.q[1] = *reinterpret_cast<const uint4*>(&Abuf[w][m][ 8 + 16 * half]);
      a1.q[0] = *reinterpret_cast<const uint4*>(&Abuf[w][m][32 + 16 * half]);
      a1.q[1] = *reinterpret_cast<const uint4*>(&Abuf[w][m][40 + 16 * half]);
    }
    __builtin_amdgcn_wave_barrier();

    #pragma unroll
    for (int s = 0; s < 4; ++s) {
      v8f aI = (v8f)biasv[s];
      v8f aF = (v8f)biasv[4 + s];
      v8f aG = (v8f)biasv[8 + s];
      v8f aO = (v8f)biasv[12 + s];
      aI = __builtin_amdgcn_wmma_f32_16x16x32_bf16(false, a0.v, false, Bop[s][0],      (short)0, aI, false, false);
      aI = __builtin_amdgcn_wmma_f32_16x16x32_bf16(false, a1.v, false, Bop[s][1],      (short)0, aI, false, false);
      aF = __builtin_amdgcn_wmma_f32_16x16x32_bf16(false, a0.v, false, Bop[4 + s][0],  (short)0, aF, false, false);
      aF = __builtin_amdgcn_wmma_f32_16x16x32_bf16(false, a1.v, false, Bop[4 + s][1],  (short)0, aF, false, false);
      aG = __builtin_amdgcn_wmma_f32_16x16x32_bf16(false, a0.v, false, Bop[8 + s][0],  (short)0, aG, false, false);
      aG = __builtin_amdgcn_wmma_f32_16x16x32_bf16(false, a1.v, false, Bop[8 + s][1],  (short)0, aG, false, false);
      aO = __builtin_amdgcn_wmma_f32_16x16x32_bf16(false, a0.v, false, Bop[12 + s][0], (short)0, aO, false, false);
      aO = __builtin_amdgcn_wmma_f32_16x16x32_bf16(false, a1.v, false, Bop[12 + s][1], (short)0, aO, false, false);

      const int  col = 16 * s + n15;       // hidden index
      const bool cv  = (col < H);
      #pragma unroll
      for (int r = 0; r < 8; ++r) {
        float ig = fsigmoid(aI[r]);
        float fg = fsigmoid(aF[r]);
        float gg = ftanh(aG[r]);
        float og = fsigmoid(aO[r]);
        float cc = fg * cs[s][r] + ig * gg;
        cs[s][r] = cc;
        float hh = og * ftanh(cc);
        if (cv) Abuf[w][r + 8 * half][col] = f2bf(hh);   // h_t -> next step's A
      }
    }
  }

  // out[b] = h_T . W_out + b_out
  if (lane < 16) {
    float acc = b_out[0];
    for (int k = 0; k < H; ++k)
      acc += bf2f(Abuf[w][lane][k]) * W_out[k];
    out[m0 + lane] = acc;
  }
}

extern "C" void kernel_launch(void* const* d_in, const int* in_sizes, int n_in,
                              void* d_out, int out_size, void* d_ws, size_t ws_size,
                              hipStream_t stream) {
  const float* seq   = (const float*)d_in[0];
  const float* W_ih  = (const float*)d_in[1];
  const float* W_hh  = (const float*)d_in[2];
  const float* b_ih  = (const float*)d_in[3];
  const float* b_hh  = (const float*)d_in[4];
  const float* W_out = (const float*)d_in[5];
  const float* b_out = (const float*)d_in[6];
  float* out = (float*)d_out;

  const int B = in_sizes[0] / (T_STEPS * 4);   // 4096
  const int blocks = B / 64;                   // 4 waves/block, 16 rows/wave
  lstm_wmma_kernel<<<blocks, 128, 0, stream>>>(seq, W_ih, W_hh, b_ih, b_hh,
                                               W_out, b_out, out);
}
